// PPAGuidedContrastiveLoss_19292993094212
// MI455X (gfx1250) — matrix operand order repaired and checked by reading, not compiled, not run on hardware
//
#include <hip/hip_runtime.h>
#include <hip/hip_bf16.h>

// ---------- types ----------
typedef __attribute__((ext_vector_type(16))) _Float16 v16h;
typedef __attribute__((ext_vector_type(8)))  _Float16 v8h;
typedef __attribute__((ext_vector_type(4)))  _Float16 v4h;
typedef __attribute__((ext_vector_type(8)))  float    v8f;

union Frag16 { v16h v; v8h h[2]; };

#define BN 4096
#define DK 1024
#define INV_TEMP (1.0f / 0.07f)
#define NEG_THRESH 0.8f   // 1 - MARGIN
#define STRIPS 4          // M-blocking: 4 x 16 = 64 rows per workgroup
#define MROWS (16 * STRIPS)

// ---------- kernel 1: row L2-normalize fp32 -> f16 ----------
__global__ void normalize_kernel(const float* __restrict__ ce,
                                 const float* __restrict__ pe,
                                 _Float16* __restrict__ ceh,
                                 _Float16* __restrict__ peh) {
    int row   = blockIdx.x & (BN - 1);
    int which = blockIdx.x >> 12;           // 0 -> ce, 1 -> pe
    const float* src = which ? pe : ce;
    _Float16*    dst = which ? peh : ceh;

    __shared__ float sred[256];
    __shared__ float sinv;

    const float4 f4 = ((const float4*)(src + (size_t)row * DK))[threadIdx.x];
    float ss = f4.x * f4.x + f4.y * f4.y + f4.z * f4.z + f4.w * f4.w;
    sred[threadIdx.x] = ss;
    __syncthreads();
    for (int s = 128; s > 0; s >>= 1) {
        if (threadIdx.x < s) sred[threadIdx.x] += sred[threadIdx.x + s];
        __syncthreads();
    }
    if (threadIdx.x == 0) {
        float nrm = sqrtf(sred[0]);
        sinv = 1.0f / fmaxf(nrm, 1e-12f);
    }
    __syncthreads();
    float inv = sinv;
    v4h o;
    o[0] = (_Float16)(f4.x * inv);
    o[1] = (_Float16)(f4.y * inv);
    o[2] = (_Float16)(f4.z * inv);
    o[3] = (_Float16)(f4.w * inv);
    ((v4h*)(dst + (size_t)row * DK))[threadIdx.x] = o;
}

// ---------- kernel 2: per-sample cell / rank / flags ----------
__global__ void flags_kernel(const int* __restrict__ gid,
                             const int* __restrict__ dt,
                             const float* __restrict__ score,
                             int* __restrict__ cellid,
                             int* __restrict__ flag,
                             int* __restrict__ csize_out) {
    int i = blockIdx.x;
    int gi = gid[i], di = dt[i];
    float si = score[i];
    int cs = 0, rk = 0;
    for (int j = threadIdx.x; j < BN; j += 256) {
        if (gid[j] == gi && dt[j] == di) {
            cs++;
            float sj = score[j];
            if (sj > si || (sj == si && j < i)) rk++;
        }
    }
    __shared__ int scs[256], srk[256];
    scs[threadIdx.x] = cs;
    srk[threadIdx.x] = rk;
    __syncthreads();
    for (int s = 128; s > 0; s >>= 1) {
        if (threadIdx.x < s) {
            scs[threadIdx.x] += scs[threadIdx.x + s];
            srk[threadIdx.x] += srk[threadIdx.x + s];
        }
        __syncthreads();
    }
    if (threadIdx.x == 0) {
        int csz = scs[0], rank = srk[0];
        int k = min(2, csz / 3 + 1);
        bool valid = csz >= 2;
        int f = 0;
        if (valid && rank < k) f |= 1;                       // positive
        if (valid && csz > k && rank >= csz - k) f |= 2;     // negative
        cellid[i]    = gi * 4 + di;
        flag[i]      = f;
        csize_out[i] = csz;
    }
}

// ---------- kernel 3: zero the accumulators (ws is poisoned) ----------
__global__ void zero_kernel(float* __restrict__ rowsum,
                            float* __restrict__ diag,
                            float* __restrict__ csum) {
    int i = blockIdx.x * blockDim.x + threadIdx.x;
    if (i < BN) { rowsum[i] = 0.0f; diag[i] = 0.0f; }
    if (i == 0) csum[0] = 0.0f;
}

// ---------- kernel 4: fused WMMA GEMMs + epilogue ----------
// grid (BN/64, 8); block 256 (8 waves).  Each WG owns a 64-row A strip in
// LDS (128 KB), staged with async global->LDS b128 copies into a
// K-block-major, XOR-swizzled layout:
//   halves addr of (row, k) = (k/32)*2048 + row*32 + swizzled(k%32)
//   16B-chunk swizzle: phys_chunk = chunk ^ ((row>>2)&3)
// -> every k-step's 8 A loads are base+immediate ds_load_b128 (no address
//    VALU in the loop) and each 16-lane half covers all 64 banks once.
// Per K-step: 4 A fragments (LDS) + 2 B fragments (global) feed 8 WMMAs.
__global__ void __launch_bounds__(256)
gemm_kernel(const _Float16* __restrict__ ceh,
            const _Float16* __restrict__ peh,
            const int* __restrict__ cellid,
            const int* __restrict__ flag,
            float* __restrict__ rowsum,
            float* __restrict__ diag,
            float* __restrict__ csum) {
    const int m0      = blockIdx.x * MROWS;
    const int colBase = blockIdx.y * 512;
    const int wave = threadIdx.x >> 5;
    const int lane = threadIdx.x & 31;
    const int n  = lane & 15;     // row-in-strip for A, col-in-tile for B/C
    const int hh = lane >> 4;     // lane-half

    __shared__ v8h AsV[MROWS * DK / 8];          // 64 x 1024 f16 = 128 KB
    __shared__ int cellRow[MROWS];
    _Float16* As = (_Float16*)AsV;

    // stage the 64-row A strip (ce_n) into LDS with async global->LDS copies,
    // writing through the K-block-major swizzled layout
    {
        const char*    gsrc = (const char*)(ceh + (size_t)m0 * DK);
        const unsigned base = (unsigned)(uintptr_t)AsV;      // LDS byte offset
        for (int t = threadIdx.x; t < MROWS * DK / 8; t += 256) {
            const int row  = t >> 7;          // 128 x 16B chunks per row
            const int kc   = t & 127;         // chunk within row
            const int blk  = kc >> 2;         // 32-K block (4 chunks each)
            const int c    = kc & 3;
            const int phys = c ^ ((row >> 2) & 3);
            unsigned      ldsAddr = base + (unsigned)(blk * 4096 + row * 64 + phys * 16);
            unsigned long gAddr   = (unsigned long)(uintptr_t)(gsrc + (size_t)t * 16);
            asm volatile("global_load_async_to_lds_b128 %0, %1, off"
                         :: "v"(ldsAddr), "v"(gAddr) : "memory");
        }
        asm volatile("s_wait_asynccnt 0x0" ::: "memory");
    }
    if (threadIdx.x < MROWS) cellRow[threadIdx.x] = cellid[m0 + threadIdx.x];
    __syncthreads();

    // per-lane swizzled base offsets (halves) for A fragment halves
    const int phys0 = hh ^ ((n >> 2) & 3);        // chunk hh     (K = hh*8..)
    const int phys1 = phys0 ^ 2;                  // chunk 2+hh   (K = 16+hh*8..)
    const _Float16* paBase0 = As + n * 32 + phys0 * 8;
    const _Float16* paBase1 = As + n * 32 + phys1 * 8;

    float csum_local = 0.0f;
    float rexp[STRIPS][8];
#pragma unroll
    for (int s = 0; s < STRIPS; ++s)
#pragma unroll
        for (int r = 0; r < 8; ++r) rexp[s][r] = 0.0f;

    for (int t = 0; t < 4; ++t) {
        const int jt   = wave + t * 8;           // tile 0..31 within chunk
        const int jcol = colBase + jt * 16 + n;  // this lane's output column
        const _Float16* prowPe = peh + (size_t)jcol * DK;
        const _Float16* prowCe = ceh + (size_t)jcol * DK;

        if (t < 3) {  // prefetch next tile's B rows into L2
            const int jn = colBase + (wave + (t + 1) * 8) * 16 + n;
            __builtin_prefetch(peh + (size_t)jn * DK, 0, 0);
            __builtin_prefetch(ceh + (size_t)jn * DK, 0, 0);
        }

        v8f accL[STRIPS] = {{}, {}, {}, {}};   // logits tiles per strip
        v8f accS[STRIPS] = {{}, {}, {}, {}};   // S tiles per strip
        const _Float16* pa0 = paBase0;
        const _Float16* pa1 = paBase1;
#pragma unroll 2
        for (int k0 = 0; k0 < DK; k0 += 32) {
            Frag16 bp, bc;
            // B 32x16 f16 layout: lane-half hh holds K = hh*8..+7 and 16+hh*8..+7, col n
            bp.h[0] = *(const v8h*)(prowPe + k0 + hh * 8);
            bp.h[1] = *(const v8h*)(prowPe + k0 + 16 + hh * 8);
            bc.h[0] = *(const v8h*)(prowCe + k0 + hh * 8);
            bc.h[1] = *(const v8h*)(prowCe + k0 + 16 + hh * 8);
#pragma unroll
            for (int s = 0; s < STRIPS; ++s) {
                Frag16 a;   // strip s row = n + 16*s -> +512 halves per strip
                a.h[0] = *(const v8h*)(pa0 + s * 512);
                a.h[1] = *(const v8h*)(pa1 + s * 512);
                accL[s] = __builtin_amdgcn_wmma_f32_16x16x32_f16(
                    false, a.v, false, bp.v, (short)0, accL[s], false, false);
                accS[s] = __builtin_amdgcn_wmma_f32_16x16x32_f16(
                    false, a.v, false, bc.v, (short)0, accS[s], false, false);
            }
            pa0 += 2048;    // next 32-K block (4 KB)
            pa1 += 2048;
        }

        const int cj = cellid[jcol];
        const int fj = flag[jcol];
#pragma unroll
        for (int s = 0; s < STRIPS; ++s) {
#pragma unroll
            for (int r = 0; r < 8; ++r) {
                const int i = m0 + s * 16 + r + 8 * hh;  // C/D: VGPR r -> M=r+8*hh
                const float lg = accL[s][r] * INV_TEMP;
                rexp[s][r] += __expf(lg);                // bounded: |lg| <= 14.3
                if (i == jcol) diag[i] = lg;
                const float sv = accS[s][r];
                if (cellRow[s * 16 + r + 8 * hh] == cj && i != jcol) {
                    if (fj & 1) csum_local += fmaxf(1.0f - sv, 0.0f);
                    if (fj & 2) csum_local += fmaxf(sv - NEG_THRESH, 0.0f);
                }
            }
        }
    }

    // reduce exp-sums across the 16 column-lanes of each half, then atomics
#pragma unroll
    for (int s = 0; s < STRIPS; ++s) {
#pragma unroll
        for (int r = 0; r < 8; ++r) {
            float v = rexp[s][r];
            v += __shfl_xor(v, 1);
            v += __shfl_xor(v, 2);
            v += __shfl_xor(v, 4);
            v += __shfl_xor(v, 8);
            if (n == 0) atomicAdd(&rowsum[m0 + s * 16 + r + 8 * hh], v);
        }
    }
    float c = csum_local;
    c += __shfl_xor(c, 1);
    c += __shfl_xor(c, 2);
    c += __shfl_xor(c, 4);
    c += __shfl_xor(c, 8);
    c += __shfl_xor(c, 16);
    if (lane == 0) atomicAdd(csum, c);
}

// ---------- kernel 5: finalize ----------
__global__ void finalize_kernel(const float* __restrict__ rowsum,
                                const float* __restrict__ diag,
                                const int* __restrict__ flag,
                                const int* __restrict__ csize,
                                const float* __restrict__ csum,
                                float* __restrict__ out) {
    __shared__ float sred[256];
    __shared__ int   cred[256];
    float s = 0.0f;
    int cnt = 0;
    for (int i = threadIdx.x; i < BN; i += 256) {
        s += logf(rowsum[i]) - diag[i];
        const int base = csize[i] - 1;
        const int f = flag[i];
        if (f & 1) cnt += base;
        if (f & 2) cnt += base;
    }
    sred[threadIdx.x] = s;
    cred[threadIdx.x] = cnt;
    __syncthreads();
    for (int st = 128; st > 0; st >>= 1) {
        if (threadIdx.x < st) {
            sred[threadIdx.x] += sred[threadIdx.x + st];
            cred[threadIdx.x] += cred[threadIdx.x + st];
        }
        __syncthreads();
    }
    if (threadIdx.x == 0) {
        const float code_ppa = sred[0] / (float)BN;
        const int   count    = cred[0];
        const float contrast = (count > 0) ? (csum[0] / (float)max(count, 1)) : 0.0f;
        const float total    = (count > 0) ? (0.6f * code_ppa + 0.4f * contrast) : code_ppa;
        out[0] = total;
        out[1] = code_ppa;
        out[2] = contrast;
    }
}

// ---------- launcher ----------
extern "C" void kernel_launch(void* const* d_in, const int* in_sizes, int n_in,
                              void* d_out, int out_size, void* d_ws, size_t ws_size,
                              hipStream_t stream) {
    const float* ce    = (const float*)d_in[0];
    const float* pe    = (const float*)d_in[1];
    const int*   gid   = (const int*)d_in[2];
    const float* score = (const float*)d_in[3];
    const int*   dt    = (const int*)d_in[4];
    // d_in[5] (is_seed) unused by the reference forward
    float* out = (float*)d_out;

    char* ws = (char*)d_ws;
    const size_t embBytes = (size_t)BN * DK * 2;   // f16
    _Float16* ceh    = (_Float16*)(ws);
    _Float16* peh    = (_Float16*)(ws + embBytes);
    int*      cellid = (int*)  (ws + 2 * embBytes);
    int*      flag   = (int*)  (ws + 2 * embBytes + 1 * (size_t)BN * 4);
    int*      csize  = (int*)  (ws + 2 * embBytes + 2 * (size_t)BN * 4);
    float*    rowsum = (float*)(ws + 2 * embBytes + 3 * (size_t)BN * 4);
    float*    diag   = (float*)(ws + 2 * embBytes + 4 * (size_t)BN * 4);
    float*    csum   = (float*)(ws + 2 * embBytes + 5 * (size_t)BN * 4);

    normalize_kernel<<<2 * BN, 256, 0, stream>>>(ce, pe, ceh, peh);
    flags_kernel<<<BN, 256, 0, stream>>>(gid, dt, score, cellid, flag, csize);
    zero_kernel<<<(BN + 255) / 256, 256, 0, stream>>>(rowsum, diag, csum);
    gemm_kernel<<<dim3(BN / 64, 8), 256, 0, stream>>>(ceh, peh, cellid, flag,
                                                      rowsum, diag, csum);
    finalize_kernel<<<1, 256, 0, stream>>>(rowsum, diag, flag, csize, csum, out);
}
